// VoxelResBackBone8xVoxelNeXt_231928234574
// MI455X (gfx1250) — compile-verified, matrix operand order checked by reading, not run on hardware
//
#include <hip/hip_runtime.h>
#include <hip/hip_bf16.h>

// ---------------------------------------------------------------------------
// VoxelResBackBone8x (VoxelNeXt) forward for MI455X / gfx1250.
// Conv layers are LDS-staged implicit-GEMM on the CDNA5 WMMA path:
//   v_wmma_f32_16x16x32_f16; NW waves per block share one 16-voxel A tile.
//   Per-tap f32 weight slabs are staged into LDS by the Tensor Data Mover
//   (tensor_load_to_lds + s_wait_tensorcnt), WMMA fragments come from LDS as
//   contiguous ds_load_b128 pairs (ISA 7.12.2 layout).
// ---------------------------------------------------------------------------

typedef __attribute__((ext_vector_type(16))) _Float16 v16h;
typedef __attribute__((ext_vector_type(8)))  _Float16 v8h;
typedef __attribute__((ext_vector_type(8)))  float    v8f;
typedef __attribute__((ext_vector_type(4)))  unsigned int u32x4;
typedef __attribute__((ext_vector_type(8)))  int      i32x8;
typedef __attribute__((ext_vector_type(4)))  int      i32x4;

#if defined(__HIP_DEVICE_COMPILE__) && __has_builtin(__builtin_amdgcn_tensor_load_to_lds)
#define HAVE_TDM 1
#else
#define HAVE_TDM 0
#endif

#define DGRID 17
#define HGRID 192
#define WGRID 192
#define CIN_V 5
#define NVOX  50000

// ---------------------------------------------------------------------------
// Scatter voxel features into dense grid (atomic add) + occupancy mask.
// ---------------------------------------------------------------------------
__global__ void scatter_kernel(const float* __restrict__ feats,
                               const int*   __restrict__ coords,
                               float* __restrict__ dense,
                               float* __restrict__ mask, int n) {
  int i = blockIdx.x * blockDim.x + threadIdx.x;
  if (i >= n) return;
  int z = coords[i * 4 + 1];
  int y = coords[i * 4 + 2];
  int x = coords[i * 4 + 3];
  size_t sp = ((size_t)z * HGRID + y) * WGRID + x;
#pragma unroll
  for (int c = 0; c < CIN_V; ++c)
    atomicAdd(&dense[sp * CIN_V + c], feats[i * CIN_V + c]);
  mask[sp] = 1.0f;
}

// ---------------------------------------------------------------------------
// LDS-staged implicit-GEMM conv (3D when KD==3, 2D when KD==1), 3x3(x3), pad 1.
// Block = NW waves; each wave owns a 16-channel slice of Cout; all waves share
// the 16-output-voxel A tile. Epilogue fuses +bias, BN, +residual, ReLU, *mask.
// ---------------------------------------------------------------------------
template <int KD, int NW>
__global__ void __launch_bounds__(32 * NW)
conv_wmma_kernel(const float* __restrict__ in, float* __restrict__ out,
                 const float* __restrict__ w, const float* __restrict__ bias,
                 const float* __restrict__ bng, const float* __restrict__ bnb,
                 const float* __restrict__ bnm, const float* __restrict__ bnv,
                 float eps,
                 const float* __restrict__ residual,
                 const float* __restrict__ mask,
                 int ID, int IH, int IW, int Cin,
                 int OD, int OH, int OW, int Cout, int stride) {
  constexpr int ECOLS = 16 * NW;       // out-channels handled per block
  const int tid    = threadIdx.x;
  const int lane   = tid & 31;
  const int waveId = tid >> 5;
  const int tileM  = blockIdx.x;       // 16 output voxels
  const int nOut   = OD * OH * OW;
  const int colBase = blockIdx.y * ECOLS;
  const int nCol    = colBase + waveId * 16 + (lane & 15);
  const int hi      = lane >> 4;       // 0 or 1

  __shared__ alignas(16) _Float16 sA[16 * 32];       // [row][k]
  __shared__ alignas(16) _Float16 sB[ECOLS * 32];    // [col][k]  (transposed)
  __shared__ alignas(16) float    sW[128 * ECOLS];   // per-tap f32 weight slab
  __shared__ int sOZ[16], sOY[16], sOX[16], sV[16];

  // Decode the 16 output positions of this tile once (kills per-lane int div).
  if (tid < 16) {
    int p = tileM * 16 + tid;
    int valid = (p < nOut) ? 1 : 0;
    int pp = valid ? p : 0;
    int oz = pp / (OH * OW);
    int r  = pp % (OH * OW);
    sOZ[tid] = oz; sOY[tid] = r / OW; sOX[tid] = r % OW; sV[tid] = valid;
  }
  __syncthreads();

  v8f acc = {};
  constexpr int AE = 512 / (32 * NW);  // sA elements per thread

  for (int tz = 0; tz < KD; ++tz) {
    for (int ty = 0; ty < 3; ++ty) {
      for (int tx = 0; tx < 3; ++tx) {
        const int tap = (tz * 3 + ty) * 3 + tx;
        const float* wp = w + (size_t)tap * Cin * Cout;
        __builtin_prefetch(wp, 0, 1);  // global_prefetch of the weight stream

        // ---- TDM: stage the f32 weight tile (Cin x ECOLS, row stride Cout)
        //      for this tap into LDS via the Tensor Data Mover. -------------
#if HAVE_TDM
        if (waveId == 0) {
          unsigned long long ga = (unsigned long long)(const char*)(wp + colBase);
          unsigned int la = (unsigned int)(unsigned long long)(void*)&sW[0];
          // D# group 0: count=1, lds_addr, global_addr[120:64], type=2.
          u32x4 g0 = {1u, la, (unsigned int)ga,
                      (unsigned int)((ga >> 32) & 0x1FFFFFFull) | (2u << 30)};
          // D# group 1: data_size=2(4B); tensor_dim0=tile_dim0=ECOLS;
          //             tensor_dim1=tile_dim1=Cin; dim0 stride=Cout.
          unsigned long long strd = (unsigned long long)Cout;
          i32x8 g1 = {
              (int)(2u << 16),
              (int)(((unsigned)ECOLS & 0xffffu) << 16),
              (int)((((unsigned)ECOLS >> 16) & 0xffffu) |
                    (((unsigned)Cin & 0xffffu) << 16)),
              (int)((((unsigned)Cin >> 16) & 0xffffu) |
                    (((unsigned)ECOLS & 0xffffu) << 16)),
              (int)((unsigned)Cin & 0xffffu),
              (int)(strd & 0xffffffffull),
              (int)((strd >> 32) & 0xffffull),
              0};
          i32x4 gz = {0, 0, 0, 0};
#if __clang_major__ >= 23
          i32x8 gz8 = {0, 0, 0, 0, 0, 0, 0, 0};
          __builtin_amdgcn_tensor_load_to_lds(g0, g1, gz, gz, gz8, 0);
#else
          __builtin_amdgcn_tensor_load_to_lds(g0, g1, gz, gz, 0);
#endif
        }
#if __has_builtin(__builtin_amdgcn_s_wait_tensorcnt)
        __builtin_amdgcn_s_wait_tensorcnt(0);
#endif
#else   // !HAVE_TDM: cooperative global fill of the weight slab
        for (int e = tid; e < Cin * ECOLS; e += 32 * NW) {
          int k = e / ECOLS, col = e % ECOLS;
          sW[e] = wp[(size_t)k * Cout + colBase + col];
        }
#endif
        __syncthreads();

        for (int c0 = 0; c0 < Cin; c0 += 32) {
          // ---- cooperative fill: sA = f16(input patch), [row][k] ----------
#pragma unroll
          for (int j = 0; j < AE; ++j) {
            int e = tid * AE + j;
            int row = e >> 5;
            int k   = e & 31;
            int kc  = c0 + k;
            int iz = sOZ[row] * stride + tz - (KD >> 1);
            int iy = sOY[row] * stride + ty - 1;
            int ix = sOX[row] * stride + tx - 1;
            bool inb = sV[row] && kc < Cin && iz >= 0 && iz < ID &&
                       iy >= 0 && iy < IH && ix >= 0 && ix < IW;
            float v = inb ? in[(((size_t)iz * IH + iy) * IW + ix) * Cin + kc]
                          : 0.0f;
            sA[e] = (_Float16)v;
          }
          // ---- fill sB = f16(sW slab), stored transposed [col][k] ---------
#pragma unroll
          for (int j = 0; j < 16; ++j) {
            int e   = tid * 16 + j;
            int k   = e / ECOLS;
            int col = e % ECOLS;
            int kc  = c0 + k;
            float v = (kc < Cin) ? sW[kc * ECOLS + col] : 0.0f;
            sB[col * 32 + k] = (_Float16)v;
          }
          __syncthreads();

          // ---- fragments from LDS: two b128 runs each (ISA 7.12.2) --------
          const int m  = lane & 15;
          const int cl = waveId * 16 + (lane & 15);
          v8h alo = *reinterpret_cast<const v8h*>(&sA[m * 32 + hi * 8]);
          v8h ahi = *reinterpret_cast<const v8h*>(&sA[m * 32 + 16 + hi * 8]);
          v8h blo = *reinterpret_cast<const v8h*>(&sB[cl * 32 + hi * 8]);
          v8h bhi = *reinterpret_cast<const v8h*>(&sB[cl * 32 + 16 + hi * 8]);
          v16h a, b;
#pragma unroll
          for (int e = 0; e < 8; ++e) {
            a[e] = alo[e]; a[e + 8] = ahi[e];
            b[e] = blo[e]; b[e + 8] = bhi[e];
          }
          acc = __builtin_amdgcn_wmma_f32_16x16x32_f16(
              /*neg_a=*/false, a, /*neg_b=*/false, b,
              /*c_mod=*/(short)0, acc, /*reuse_a=*/false, /*reuse_b=*/false);
          __syncthreads();
        }
      }
    }
  }

  // Fused epilogue: bias -> BN -> residual -> ReLU -> mask.
  const float scale = bng[nCol] * rsqrtf(bnv[nCol] + eps);
  const float shift = bnb[nCol] - bnm[nCol] * scale;
  const float bs = bias ? bias[nCol] : 0.0f;
#pragma unroll
  for (int v = 0; v < 8; ++v) {
    const int m  = v + (hi ? 8 : 0);
    const int pp = tileM * 16 + m;
    if (pp < nOut) {
      float val = (acc[v] + bs) * scale + shift;
      if (residual) val += residual[(size_t)pp * Cout + nCol];
      val = fmaxf(val, 0.0f);
      if (mask) val *= mask[pp];
      out[(size_t)pp * Cout + nCol] = val;
    }
  }
}

// ---------------------------------------------------------------------------
// Mask downsample: m2 = (conv3d(mask, ones3^3, stride 2, pad 1) > 0)
// ---------------------------------------------------------------------------
__global__ void mask_down_kernel(const float* __restrict__ mi,
                                 float* __restrict__ mo,
                                 int ID, int IH, int IW,
                                 int OD, int OH, int OW) {
  int p = blockIdx.x * blockDim.x + threadIdx.x;
  if (p >= OD * OH * OW) return;
  int oz = p / (OH * OW);
  int r  = p % (OH * OW);
  int oy = r / OW, ox = r % OW;
  float s = 0.0f;
  for (int tz = 0; tz < 3; ++tz) {
    int iz = oz * 2 + tz - 1;
    if (iz < 0 || iz >= ID) continue;
    for (int ty = 0; ty < 3; ++ty) {
      int iy = oy * 2 + ty - 1;
      if (iy < 0 || iy >= IH) continue;
      for (int tx = 0; tx < 3; ++tx) {
        int ix = ox * 2 + tx - 1;
        if (ix < 0 || ix >= IW) continue;
        s += mi[((size_t)iz * IH + iy) * IW + ix];
      }
    }
  }
  mo[p] = (s > 0.0f) ? 1.0f : 0.0f;
}

// ---------------------------------------------------------------------------
// Multi-scale combine (x4 + up(x5,2) + up(x6,4)), BEV depth-sum, BEV mask-max.
// x4:(3,24,24,128) x5:(2,12,12,128) x6:(1,6,6,128).
// ---------------------------------------------------------------------------
__global__ void bev_kernel(const float* __restrict__ x4,
                           const float* __restrict__ x5,
                           const float* __restrict__ x6,
                           const float* __restrict__ m4,
                           const float* __restrict__ m5,
                           const float* __restrict__ m6,
                           float* __restrict__ bev,
                           float* __restrict__ bmask,
                           int D4, int Hh, int Ww, int C) {
  int idx = blockIdx.x * blockDim.x + threadIdx.x;
  int total = Hh * Ww * C;
  if (idx >= total) return;
  int c = idx % C;
  int rest = idx / C;
  int x = rest % Ww, y = rest / Ww;
  float s = 0.0f, mm = 0.0f;
  for (int d = 0; d < D4; ++d) {
    float v = x4[(((size_t)d * Hh + y) * Ww + x) * C + c];
    float m = m4[((size_t)d * Hh + y) * Ww + x];
    if ((d & 1) == 0 && (y & 1) == 0 && (x & 1) == 0) {
      int d2 = d >> 1, y2 = y >> 1, x2 = x >> 1;
      v += x5[(((size_t)d2 * (Hh >> 1) + y2) * (Ww >> 1) + x2) * C + c];
      m = fmaxf(m, m5[((size_t)d2 * (Hh >> 1) + y2) * (Ww >> 1) + x2]);
    }
    if ((d & 3) == 0 && (y & 3) == 0 && (x & 3) == 0) {
      int d4 = d >> 2, y4 = y >> 2, x4i = x >> 2;
      v += x6[(((size_t)d4 * (Hh >> 2) + y4) * (Ww >> 2) + x4i) * C + c];
      m = fmaxf(m, m6[((size_t)d4 * (Hh >> 2) + y4) * (Ww >> 2) + x4i]);
    }
    s += v;
    mm = fmaxf(mm, m);
  }
  bev[((size_t)y * Ww + x) * C + c] = s;
  if (c == 0) bmask[(size_t)y * Ww + x] = mm;
}

// mo = (conv2d(bmask, ones3x3, pad 1) > 0)
__global__ void dilate2d_kernel(const float* __restrict__ bm,
                                float* __restrict__ mo, int Hh, int Ww) {
  int p = blockIdx.x * blockDim.x + threadIdx.x;
  if (p >= Hh * Ww) return;
  int y = p / Ww, x = p % Ww;
  float s = 0.0f;
  for (int ty = -1; ty <= 1; ++ty) {
    int iy = y + ty;
    if (iy < 0 || iy >= Hh) continue;
    for (int tx = -1; tx <= 1; ++tx) {
      int ix = x + tx;
      if (ix < 0 || ix >= Ww) continue;
      s += bm[iy * Ww + ix];
    }
  }
  mo[p] = (s > 0.0f) ? 1.0f : 0.0f;
}

// ---------------------------------------------------------------------------
// Host-side parameter plumbing (params pytree flattened: top-level dict in
// insertion order, nested dicts in sorted-key order).
// ---------------------------------------------------------------------------
struct BNP { const float *b, *g, *m, *v; };
struct BlkP { const float *b1, *b2; BNP bn1, bn2; const float *w1, *w2; };
struct StageP { BlkP blk[2]; BNP bn; const float* w; };

static inline BNP getBN(void* const* d, int& i) {
  BNP p;
  p.b = (const float*)d[i++]; p.g = (const float*)d[i++];
  p.m = (const float*)d[i++]; p.v = (const float*)d[i++];
  return p;
}
static inline BlkP getBlk(void* const* d, int& i) {
  BlkP p;
  p.b1 = (const float*)d[i++]; p.b2 = (const float*)d[i++];
  p.bn1 = getBN(d, i); p.bn2 = getBN(d, i);
  p.w1 = (const float*)d[i++]; p.w2 = (const float*)d[i++];
  return p;
}
static inline StageP getStage(void* const* d, int& i) {
  StageP s;
  s.blk[0] = getBlk(d, i); s.blk[1] = getBlk(d, i);
  s.bn = getBN(d, i); s.w = (const float*)d[i++];
  return s;
}

static void conv3(hipStream_t st, const float* in, float* out, const float* w,
                  const float* bias, const BNP& bn, float eps,
                  const float* res, const float* mask,
                  int ID, int IH, int IW, int Cin,
                  int OD, int OH, int OW, int Cout, int stride) {
  unsigned gm = (unsigned)((OD * OH * OW + 15) / 16);
  if (Cout >= 64) {
    dim3 grid(gm, (unsigned)(Cout / 64));
    conv_wmma_kernel<3, 4><<<grid, dim3(128), 0, st>>>(
        in, out, w, bias, bn.g, bn.b, bn.m, bn.v, eps, res, mask,
        ID, IH, IW, Cin, OD, OH, OW, Cout, stride);
  } else if (Cout == 32) {
    dim3 grid(gm, 1u);
    conv_wmma_kernel<3, 2><<<grid, dim3(64), 0, st>>>(
        in, out, w, bias, bn.g, bn.b, bn.m, bn.v, eps, res, mask,
        ID, IH, IW, Cin, OD, OH, OW, Cout, stride);
  } else {
    dim3 grid(gm, 1u);
    conv_wmma_kernel<3, 1><<<grid, dim3(32), 0, st>>>(
        in, out, w, bias, bn.g, bn.b, bn.m, bn.v, eps, res, mask,
        ID, IH, IW, Cin, OD, OH, OW, Cout, stride);
  }
}

static void conv2(hipStream_t st, const float* in, float* out, const float* w,
                  const float* bias, const BNP& bn, float eps,
                  const float* res, const float* mask,
                  int IH, int IW, int Cin, int OH, int OW, int Cout) {
  dim3 grid((unsigned)((OH * OW + 15) / 16), (unsigned)(Cout / 64));
  conv_wmma_kernel<1, 4><<<grid, dim3(128), 0, st>>>(
      in, out, w, bias, bn.g, bn.b, bn.m, bn.v, eps, res, mask,
      1, IH, IW, Cin, 1, OH, OW, Cout, 1);
}

// Residual block: x -> tmp (conv1) -> x (conv2 + residual), x stays in place.
static void runBlock(hipStream_t st, float* x, float* tmp, const BlkP& bp,
                     const float* mask, int Dd, int Hh, int Ww, int C) {
  conv3(st, x, tmp, bp.w1, bp.b1, bp.bn1, 1e-3f, nullptr, mask,
        Dd, Hh, Ww, C, Dd, Hh, Ww, C, 1);
  conv3(st, tmp, x, bp.w2, bp.b2, bp.bn2, 1e-3f, x, mask,
        Dd, Hh, Ww, C, Dd, Hh, Ww, C, 1);
}

// Down stage: result lands in xout (the "other" buffer).
static void runStage(hipStream_t st, float* xin, float* xout, const StageP& s,
                     const float* maskIn, float* maskOut,
                     int ID, int IH, int IW, int Cin,
                     int OD, int OH, int OW, int Cout) {
  int n = OD * OH * OW;
  mask_down_kernel<<<(n + 255) / 256, 256, 0, st>>>(maskIn, maskOut,
                                                    ID, IH, IW, OD, OH, OW);
  conv3(st, xin, xout, s.w, nullptr, s.bn, 1e-3f, nullptr, maskOut,
        ID, IH, IW, Cin, OD, OH, OW, Cout, 2);
  runBlock(st, xout, xin, s.blk[0], maskOut, OD, OH, OW, Cout);
  runBlock(st, xout, xin, s.blk[1], maskOut, OD, OH, OW, Cout);
}

extern "C" void kernel_launch(void* const* d_in, const int* in_sizes, int n_in,
                              void* d_out, int out_size, void* d_ws,
                              size_t ws_size, hipStream_t stream) {
  (void)in_sizes; (void)n_in; (void)out_size; (void)ws_size;

  const float* feats  = (const float*)d_in[0];
  const int*   coords = (const int*)d_in[1];

  int i = 2;
  BlkP c1a = getBlk(d_in, i);
  BlkP c1b = getBlk(d_in, i);
  StageP s2 = getStage(d_in, i);
  StageP s3 = getStage(d_in, i);
  StageP s4 = getStage(d_in, i);
  StageP s5 = getStage(d_in, i);
  StageP s6 = getStage(d_in, i);
  BNP inBN = getBN(d_in, i);  const float* inW  = (const float*)d_in[i++];
  BNP outBN = getBN(d_in, i); const float* outW = (const float*)d_in[i++];
  const float* shB = (const float*)d_in[i++];
  BNP shBN = getBN(d_in, i);  const float* shW  = (const float*)d_in[i++];

  // ---- workspace carve -----------------------------------------------------
  char* ws = (char*)d_ws;
  size_t off = 0;
  auto alloc = [&](size_t nflt) -> float* {
    float* p = (float*)(ws + off);
    off += ((nflt * sizeof(float) + 255) / 256) * 256;
    return p;
  };
  const size_t nsp0 = (size_t)DGRID * HGRID * WGRID;       // 626688
  float* dense = alloc(nsp0 * CIN_V);
  float* mask0 = alloc(nsp0);
  float* A     = alloc(nsp0 * 16);                         // ping
  float* B     = alloc(nsp0 * 16);                         // pong
  float* m2    = alloc((size_t)9 * 96 * 96);
  float* m3    = alloc((size_t)5 * 48 * 48);
  float* m4    = alloc((size_t)3 * 24 * 24);
  float* m5    = alloc((size_t)2 * 12 * 12);
  float* m6    = alloc((size_t)1 * 6 * 6);
  float* x4    = alloc((size_t)3 * 24 * 24 * 128);
  float* x5    = alloc((size_t)2 * 12 * 12 * 128);
  float* x6    = alloc((size_t)1 * 6 * 6 * 128);
  float* bev   = alloc((size_t)24 * 24 * 128);
  float* bmask = alloc((size_t)24 * 24);
  float* mo    = alloc((size_t)24 * 24);
  float* hmid  = alloc((size_t)24 * 24 * 128);

  // ---- scatter -------------------------------------------------------------
  hipMemsetAsync(dense, 0, nsp0 * CIN_V * sizeof(float), stream);
  hipMemsetAsync(mask0, 0, nsp0 * sizeof(float), stream);
  scatter_kernel<<<(NVOX + 255) / 256, 256, 0, stream>>>(feats, coords, dense,
                                                         mask0, NVOX);

  // ---- conv_input + conv1 blocks (full res, 16 ch), x lives in A ----------
  conv3(stream, dense, A, inW, nullptr, inBN, 1e-3f, nullptr, mask0,
        DGRID, HGRID, WGRID, CIN_V, DGRID, HGRID, WGRID, 16, 1);
  runBlock(stream, A, B, c1a, mask0, DGRID, HGRID, WGRID, 16);
  runBlock(stream, A, B, c1b, mask0, DGRID, HGRID, WGRID, 16);

  // ---- down stages (ping-pong A<->B) --------------------------------------
  runStage(stream, A, B, s2, mask0, m2, 17, 192, 192, 16,  9, 96, 96, 32);
  runStage(stream, B, A, s3, m2,   m3,  9,  96,  96, 32,  5, 48, 48, 64);
  runStage(stream, A, B, s4, m3,   m4,  5,  48,  48, 64,  3, 24, 24, 128);
  hipMemcpyAsync(x4, B, (size_t)3 * 24 * 24 * 128 * sizeof(float),
                 hipMemcpyDeviceToDevice, stream);
  runStage(stream, B, A, s5, m4,   m5,  3,  24,  24, 128, 2, 12, 12, 128);
  hipMemcpyAsync(x5, A, (size_t)2 * 12 * 12 * 128 * sizeof(float),
                 hipMemcpyDeviceToDevice, stream);
  runStage(stream, A, B, s6, m5,   m6,  2,  12,  12, 128, 1,  6,  6, 128);
  hipMemcpyAsync(x6, B, (size_t)1 * 6 * 6 * 128 * sizeof(float),
                 hipMemcpyDeviceToDevice, stream);

  // ---- multi-scale combine + BEV ------------------------------------------
  {
    int total = 24 * 24 * 128;
    bev_kernel<<<(total + 255) / 256, 256, 0, stream>>>(
        x4, x5, x6, m4, m5, m6, bev, bmask, 3, 24, 24, 128);
    dilate2d_kernel<<<(24 * 24 + 255) / 256, 256, 0, stream>>>(bmask, mo, 24, 24);
  }

  // ---- BEV head: conv_out then shared, final into d_out -------------------
  conv2(stream, bev, hmid, outW, nullptr, outBN, 1e-3f, nullptr, mo,
        24, 24, 128, 24, 24, 128);
  conv2(stream, hmid, (float*)d_out, shW, shB, shBN, 1e-5f, nullptr, mo,
        24, 24, 128, 24, 24, 128);
}
